// CausalMultiHeadSelfAttention_55018531062302
// MI455X (gfx1250) — matrix operand order repaired and checked by reading, not compiled, not run on hardware
//
#include <hip/hip_runtime.h>
#include <hip/hip_bf16.h>

typedef __attribute__((ext_vector_type(16))) __bf16 v16bf;
typedef __attribute__((ext_vector_type(8)))  __bf16 v8bf;
typedef __attribute__((ext_vector_type(8)))  float  v8f;

#define D_MODEL   1024
#define NUM_HEADS 16
#define D_K       64
#define SEQ       2048
#define BATCH     2
#define MROWS     (BATCH * SEQ)   // 4096 total rows

// fp32 -> bf16, round-to-nearest-even
__device__ __forceinline__ __bf16 f2bf(float f) {
  unsigned u = __builtin_bit_cast(unsigned, f);
  unsigned r = (u + 0x7FFFu + ((u >> 16) & 1u)) >> 16;
  unsigned short s = (unsigned short)r;
  return __builtin_bit_cast(__bf16, s);
}

__global__ void cvt_f32_to_bf16(const float* __restrict__ in, __bf16* __restrict__ out, int n) {
  int i = blockIdx.x * blockDim.x + threadIdx.x;
  int stride = gridDim.x * blockDim.x;
  for (; i < n; i += stride) out[i] = f2bf(in[i]);
}

// ---------------------------------------------------------------------------
// C[m,n] = sum_d X[m,d] * W[n,d]   (i.e. X @ W^T), M=4096, N=K=1024, bf16 WMMA.
// One wave computes a 32x64 tile: 2 A-fragments (M-blocking) x 4 B-fragments
// (N-blocking) -> 8 WMMAs per 32-wide K step, ~0.75 KB L2 traffic per WMMA.
// mode 0: apply RoPE, store bf16 to out[b,h,s,d]           (Q and K)
// mode 2: store bf16 transposed to out[b,h,d,s]            (V)
// mode 3: store fp32 row-major [m, n]                      (final projection)
// ---------------------------------------------------------------------------
__global__ void gemm_xwt(const __bf16* __restrict__ X, const __bf16* __restrict__ W,
                         __bf16* __restrict__ outb, float* __restrict__ outf, int mode)
{
  const int lane  = threadIdx.x & 31;
  const int wid   = (blockIdx.x * blockDim.x + threadIdx.x) >> 5;
  const int NBLK  = D_MODEL / 64;            // 16 n-blocks of width 64
  const int mTile = wid / NBLK;              // 128 m-tiles of height 32
  const int nBlk  = wid - mTile * NBLK;
  const int m0    = mTile * 32;
  const int n0    = nBlk * 64;
  const int half  = lane >> 4;               // 0: lanes 0-15, 1: lanes 16-31
  const int lq    = lane & 15;

  const __bf16* xrow0 = X + (size_t)(m0 + lq) * D_MODEL;
  const __bf16* xrow1 = xrow0 + (size_t)16 * D_MODEL;
  const __bf16* wrow0 = W + (size_t)(n0 + lq) * D_MODEL;

  v8f acc[2][4] = {};

  for (int k0 = 0; k0 < D_MODEL; k0 += 32) {
    // A fragments (16x32 bf16): lane holds one row; half-dependent K chunks
    v16bf a0, a1;
    {
      v8bf lo = *(const v8bf*)(xrow0 + k0 + 16 * half);
      v8bf hi = *(const v8bf*)(xrow0 + k0 + 16 * half + 8);
      #pragma unroll
      for (int j = 0; j < 8; ++j) { a0[j] = lo[j]; a0[j + 8] = hi[j]; }
    }
    {
      v8bf lo = *(const v8bf*)(xrow1 + k0 + 16 * half);
      v8bf hi = *(const v8bf*)(xrow1 + k0 + 16 * half + 8);
      #pragma unroll
      for (int j = 0; j < 8; ++j) { a1[j] = lo[j]; a1[j + 8] = hi[j]; }
    }
    #pragma unroll
    for (int t = 0; t < 4; ++t) {
      // B fragment (32x16): lane = output col, 16 contiguous K values per half
      v16bf b = *(const v16bf*)(wrow0 + (size_t)t * 16 * D_MODEL + k0 + 16 * half);
      acc[0][t] = __builtin_amdgcn_wmma_f32_16x16x32_bf16(false, a0, false, b,
                                                          (short)0, acc[0][t], false, false);
      acc[1][t] = __builtin_amdgcn_wmma_f32_16x16x32_bf16(false, a1, false, b,
                                                          (short)0, acc[1][t], false, false);
    }
  }

  if (mode == 3) {                               // fp32 row-major output
    #pragma unroll
    for (int r = 0; r < 2; ++r)
      #pragma unroll
      for (int t = 0; t < 4; ++t) {
        int n = n0 + 16 * t + lq;
        #pragma unroll
        for (int i = 0; i < 8; ++i) {
          int row = m0 + 16 * r + i + 8 * half;  // D-frag: M = i + 8*half, N = lq
          outf[(size_t)row * D_MODEL + n] = acc[r][t][i];
        }
      }
    return;
  }

  if (mode == 2) {                               // V -> [b,h,d,s] bf16
    #pragma unroll
    for (int r = 0; r < 2; ++r)
      #pragma unroll
      for (int t = 0; t < 4; ++t) {
        int n = n0 + 16 * t + lq;
        int h = n >> 6, d = n & 63;
        #pragma unroll
        for (int i = 0; i < 8; ++i) {
          int row = m0 + 16 * r + i + 8 * half;
          int s = row & (SEQ - 1), bb = row >> 11;
          outb[(((size_t)(bb * NUM_HEADS + h) * D_K + d) << 11) + s] = f2bf(acc[r][t][i]);
        }
      }
    return;
  }

  // mode 0: RoPE, then [b,h,s,d] bf16. Pair partner (d^1) sits in lane^1.
  #pragma unroll
  for (int r = 0; r < 2; ++r)
    #pragma unroll
    for (int t = 0; t < 4; ++t) {
      int n = n0 + 16 * t + lq;
      int h = n >> 6, d = n & 63;
      int p = d >> 1;
      float inv_freq = __expf(-(float)p * 0.2878231366242557f); // ln(10000)/32
      #pragma unroll
      for (int i = 0; i < 8; ++i) {
        int row = m0 + 16 * r + i + 8 * half;
        int s = row & (SEQ - 1), bb = row >> 11;
        float cs, sn;
        __sincosf((float)s * inv_freq, &cs, &sn);
        float v = acc[r][t][i];
        float o = __shfl_xor(v, 1, 32);
        float rr = (d & 1) ? (o * sn + v * cs) : (v * cs - o * sn);
        outb[(((size_t)(bb * NUM_HEADS + h) * SEQ + s) << 6) + d] = f2bf(rr);
      }
    }
}

// ---------------------------------------------------------------------------
// Flash attention: one wave per (b, h, 16-query tile).
// Computes S^T = K_tile(16xD_K) @ Q_tile^T so each lane owns one query column;
// online softmax with a single shfl_xor(16) cross-half reduction; O^T
// accumulated via V^T(16x32) @ P^T(32x16) WMMAs with flash rescaling.
// ---------------------------------------------------------------------------
__global__ void attn_flash(const __bf16* __restrict__ Qr, const __bf16* __restrict__ Kr,
                           const __bf16* __restrict__ Vt, __bf16* __restrict__ AO)
{
  const int lane  = threadIdx.x & 31;
  const int wid   = (blockIdx.x * blockDim.x + threadIdx.x) >> 5;
  const int qTile = wid & 127;
  const int h     = (wid >> 7) & 15;
  const int b     = wid >> 11;
  const int q0    = qTile << 4;
  const int half  = lane >> 4;
  const int lq    = lane & 15;
  const size_t bh = (size_t)(b * NUM_HEADS + h);

  // Q^T B-fragments (d is the K-axis): lane = query, contiguous d per half
  const __bf16* qrow = Qr + (bh * SEQ + q0 + lq) * D_K;
  v16bf qf[2];
  #pragma unroll
  for (int c = 0; c < 2; ++c) qf[c] = *(const v16bf*)(qrow + 32 * c + 16 * half);

  const __bf16* Kb = Kr + bh * SEQ * D_K;
  const __bf16* Vb = Vt + bh * D_K * SEQ;

  v8f o[4] = {};
  float m_run = -1e30f, l_run = 0.f;
  const int kend = q0 + 15;

  for (int kc = 0; kc <= kend; kc += 32) {       // wave-uniform trip count
    v8f st[2];
    #pragma unroll
    for (int t = 0; t < 2; ++t) {
      int kb = kc + 16 * t;
      if (kb <= kend) {                          // wave-uniform branch
        v8f sacc = {};
        const __bf16* krow = Kb + (size_t)(kb + lq) * D_K;
        #pragma unroll
        for (int c = 0; c < 2; ++c) {
          v8bf klo = *(const v8bf*)(krow + 32 * c + 16 * half);
          v8bf khi = *(const v8bf*)(krow + 32 * c + 16 * half + 8);
          v16bf ka;
          #pragma unroll
          for (int j = 0; j < 8; ++j) { ka[j] = klo[j]; ka[j + 8] = khi[j]; }
          sacc = __builtin_amdgcn_wmma_f32_16x16x32_bf16(false, ka, false, qf[c],
                                                         (short)0, sacc, false, false);
        }
        #pragma unroll
        for (int i = 0; i < 8; ++i) {            // S^T frag: M=key=i+8*half, N=query=lq
          int k = kb + i + 8 * half;
          float sv = sacc[i] * 0.125f;           // 1/sqrt(64)
          st[t][i] = (k > q0 + lq) ? -1e30f : sv;
        }
      } else {
        #pragma unroll
        for (int i = 0; i < 8; ++i) st[t][i] = -1e30f;
      }
    }

    // online softmax per query (lane + its half-partner hold the 16 keys each)
    float cmax = -1e30f;
    #pragma unroll
    for (int i = 0; i < 8; ++i) cmax = fmaxf(cmax, fmaxf(st[0][i], st[1][i]));
    cmax = fmaxf(cmax, __shfl_xor(cmax, 16, 32));
    float m_new = fmaxf(m_run, cmax);
    float corr  = __expf(m_run - m_new);
    float e0[8], e1[8], csum = 0.f;
    #pragma unroll
    for (int i = 0; i < 8; ++i) {
      e0[i] = __expf(st[0][i] - m_new);
      e1[i] = __expf(st[1][i] - m_new);
      csum += e0[i] + e1[i];
    }
    csum += __shfl_xor(csum, 16, 32);
    l_run = l_run * corr + csum;
    m_run = m_new;

    // Build P^T B-fragment (32 keys on K-axis): half-swap shuffles
    v16bf pf;
    #pragma unroll
    for (int i = 0; i < 8; ++i) {
      float o0 = __shfl_xor(e0[i], 16, 32);
      float o1 = __shfl_xor(e1[i], 16, 32);
      if (half == 0) { pf[i] = f2bf(e0[i]); pf[i + 8] = f2bf(o0); }
      else           { pf[i] = f2bf(o1);   pf[i + 8] = f2bf(e1[i]); }
    }

    // O^T += V^T @ P^T  (4 d-chunks of 16), rescale accumulators first
    #pragma unroll
    for (int c = 0; c < 4; ++c) {
      const __bf16* vrow = Vb + (size_t)(16 * c + lq) * SEQ + kc;  // lane = d_local
      v8bf vlo = *(const v8bf*)(vrow + 16 * half);
      v8bf vhi = *(const v8bf*)(vrow + 16 * half + 8);
      v16bf va;
      #pragma unroll
      for (int j = 0; j < 8; ++j) { va[j] = vlo[j]; va[j + 8] = vhi[j]; }
      #pragma unroll
      for (int i = 0; i < 8; ++i) o[c][i] *= corr;
      o[c] = __builtin_amdgcn_wmma_f32_16x16x32_bf16(false, va, false, pf,
                                                     (short)0, o[c], false, false);
    }
  }

  float inv_l = 1.f / l_run;
  int s = q0 + lq;
  __bf16* aorow = AO + (size_t)(b * SEQ + s) * D_MODEL + h * D_K;
  #pragma unroll
  for (int c = 0; c < 4; ++c)
    #pragma unroll
    for (int i = 0; i < 8; ++i)
      aorow[16 * c + i + 8 * half] = f2bf(o[c][i] * inv_l);
}

extern "C" void kernel_launch(void* const* d_in, const int* in_sizes, int n_in,
                              void* d_out, int out_size, void* d_ws, size_t ws_size,
                              hipStream_t stream) {
  (void)in_sizes; (void)n_in; (void)out_size; (void)ws_size;
  const float* x  = (const float*)d_in[0];
  const float* Wq = (const float*)d_in[1];
  const float* Wk = (const float*)d_in[2];
  const float* Wv = (const float*)d_in[3];
  const float* Wo = (const float*)d_in[4];
  float* out = (float*)d_out;

  char* ws = (char*)d_ws;                         // 48 MB used, MB-aligned slots
  __bf16* Xb  = (__bf16*)(ws);                    // [4096,1024]      8 MB
  __bf16* Wqb = (__bf16*)(ws + (8u  << 20));      // [1024,1024]      2 MB each
  __bf16* Wkb = (__bf16*)(ws + (10u << 20));
  __bf16* Wvb = (__bf16*)(ws + (12u << 20));
  __bf16* Wob = (__bf16*)(ws + (14u << 20));
  __bf16* Qr  = (__bf16*)(ws + (16u << 20));      // [b,h,s,d]        8 MB
  __bf16* Kr  = (__bf16*)(ws + (24u << 20));      // [b,h,s,d]        8 MB
  __bf16* Vt  = (__bf16*)(ws + (32u << 20));      // [b,h,d,s]        8 MB
  __bf16* AO  = (__bf16*)(ws + (40u << 20));      // [b,s,e]          8 MB

  cvt_f32_to_bf16<<<1024, 256, 0, stream>>>(x,  Xb,  MROWS * D_MODEL);
  cvt_f32_to_bf16<<<512,  256, 0, stream>>>(Wq, Wqb, D_MODEL * D_MODEL);
  cvt_f32_to_bf16<<<512,  256, 0, stream>>>(Wk, Wkb, D_MODEL * D_MODEL);
  cvt_f32_to_bf16<<<512,  256, 0, stream>>>(Wv, Wvb, D_MODEL * D_MODEL);
  cvt_f32_to_bf16<<<512,  256, 0, stream>>>(Wo, Wob, D_MODEL * D_MODEL);

  // 128 mTiles * 16 nBlocks = 2048 waves -> 256 blocks of 256 threads (8 waves)
  gemm_xwt<<<256, 256, 0, stream>>>(Xb, Wqb, Qr, nullptr, 0);   // Q + RoPE
  gemm_xwt<<<256, 256, 0, stream>>>(Xb, Wkb, Kr, nullptr, 0);   // K + RoPE
  gemm_xwt<<<256, 256, 0, stream>>>(Xb, Wvb, Vt, nullptr, 2);   // V transposed

  // 2*16*128 = 4096 waves -> 512 blocks
  attn_flash<<<512, 256, 0, stream>>>(Qr, Kr, Vt, AO);

  gemm_xwt<<<256, 256, 0, stream>>>(AO, Wob, nullptr, out, 3);  // out = AO @ Wo^T (fp32)
}
